// MoE_48919677501987
// MI455X (gfx1250) — matrix operand order repaired and checked by reading, not compiled, not run on hardware
//
#include <hip/hip_runtime.h>
#include <hip/hip_bf16.h>

// Problem constants (from the reference)
#define D_DIM 2048
#define E_N   16
#define H_DIM 1024
#define TOPK  4
#define B_N   4096

// Expert-kernel tiling
#define TM    64                 // tokens per block tile
#define TN    128                // output columns per block tile
#define TK    64                 // K chunk per stage
#define TILES (B_N / TM)         // 64 M-tiles per expert (early-exit past count)

typedef __attribute__((ext_vector_type(16))) _Float16 v16h;
typedef __attribute__((ext_vector_type(8)))  _Float16 v8h;
typedef __attribute__((ext_vector_type(8)))  float    v8f;

static __device__ __forceinline__ v16h cat8(v8h a, v8h b) {
  return __builtin_shufflevector(a, b, 0,1,2,3,4,5,6,7,8,9,10,11,12,13,14,15);
}

// ---------------------------------------------------------------------------
// Kernel 0: zero the per-expert slot counters
// ---------------------------------------------------------------------------
__global__ void moe_zero_counts(int* __restrict__ counts) {
  if (threadIdx.x < E_N) counts[threadIdx.x] = 0;
}

// ---------------------------------------------------------------------------
// Kernel 1: router.  logits = x @ Wr + br ; top-4 ; softmax ; scatter to
// per-expert buckets (tokens[], gates[]) with atomic slot allocation.
// One block handles 16 tokens; thread (t,e) = (tid/16, tid%16).
// ---------------------------------------------------------------------------
__global__ __launch_bounds__(256)
void moe_router(const float* __restrict__ x,
                const float* __restrict__ Wr,
                const float* __restrict__ br,
                int*   __restrict__ counts,
                int*   __restrict__ tokens,
                float* __restrict__ gates) {
  __shared__ float xs[16][256];   // 16 KB
  __shared__ float ws[256][E_N];  // 16 KB
  __shared__ float lg[16][E_N];

  const int tid = threadIdx.x;
  const int t   = tid >> 4;       // local token 0..15
  const int e   = tid & 15;       // expert 0..15
  const int b0  = blockIdx.x * 16;

  float acc = br[e];
  for (int d0 = 0; d0 < D_DIM; d0 += 256) {
    { // stage 16x256 slab of x
      int r  = tid >> 4;
      int cb = (tid & 15) * 16;
      const float4* src = (const float4*)(x + (size_t)(b0 + r) * D_DIM + d0 + cb);
      float4* dst = (float4*)&xs[r][cb];
      #pragma unroll
      for (int q = 0; q < 4; ++q) dst[q] = src[q];
    }
    { // stage 256x16 slab of Wr (row-contiguous, 64B per row)
      const float4* src = (const float4*)(Wr + (size_t)(d0 + tid) * E_N);
      float4* dst = (float4*)&ws[tid][0];
      #pragma unroll
      for (int q = 0; q < 4; ++q) dst[q] = src[q];
    }
    __syncthreads();
    #pragma unroll 8
    for (int k = 0; k < 256; ++k) acc += xs[t][k] * ws[k][e];
    __syncthreads();
  }
  lg[t][e] = acc;
  __syncthreads();

  if (e == 0) {
    float v[E_N];
    #pragma unroll
    for (int i = 0; i < E_N; ++i) v[i] = lg[t][i];
    int   bi[TOPK];
    float bv[TOPK];
    #pragma unroll
    for (int k = 0; k < TOPK; ++k) {
      float m = -3.4e38f; int mi = 0;
      #pragma unroll
      for (int i = 0; i < E_N; ++i) { if (v[i] > m) { m = v[i]; mi = i; } }
      bv[k] = m; bi[k] = mi; v[mi] = -3.4e38f;
    }
    const float mx = bv[0];
    float ex[TOPK]; float s = 0.f;
    #pragma unroll
    for (int k = 0; k < TOPK; ++k) { ex[k] = __expf(bv[k] - mx); s += ex[k]; }
    const float inv = 1.f / s;
    const int b = b0 + t;
    #pragma unroll
    for (int k = 0; k < TOPK; ++k) {
      int exp = bi[k];
      int slot = atomicAdd(&counts[exp], 1);
      tokens[exp * B_N + slot] = b;
      gates [exp * B_N + slot] = ex[k] * inv;
    }
  }
}

// ---------------------------------------------------------------------------
// Kernel 2: residual init, out = x (float4 vectorized)
// ---------------------------------------------------------------------------
__global__ void moe_residual(const float* __restrict__ x,
                             float* __restrict__ out, int n4) {
  int i = blockIdx.x * blockDim.x + threadIdx.x;
  if (i < n4) ((float4*)out)[i] = ((const float4*)x)[i];
}

// ---------------------------------------------------------------------------
// Kernel 3: fused expert FFN.  grid = (TILES, E).  256 threads = 8 waves.
// Block tile: TM x TN; wave grid 2(M) x 4(N); per-wave 2x2 16x16 WMMA frags.
// Phase A: h = relu(Xg @ W1[e] + b1[e])  -> kept in LDS (f16, 64x1024)
// Phase B: y = h @ W2[e] + b2[e]; out[tok] += gate * y  (f32 atomics)
// ---------------------------------------------------------------------------
__global__ __launch_bounds__(256)
void moe_expert(const float* __restrict__ x,
                const float* __restrict__ W1,
                const float* __restrict__ b1,
                const float* __restrict__ W2,
                const float* __restrict__ b2,
                const int*   __restrict__ counts,
                const int*   __restrict__ tokens,
                const float* __restrict__ gates,
                float*       __restrict__ out) {
  __shared__ _Float16 sH[TM][H_DIM + 8];  // 64x1032 f16 = 129.0 KB (16B-aligned rows)
  __shared__ _Float16 sA[TM][TK + 8];     // 64x72  f16 =   9.0 KB
  __shared__ _Float16 sB[TN][TK + 8];     // 128x72 f16 =  18.0 KB (B staged transposed [n][k])
  __shared__ int   sTok[TM];
  __shared__ float sGate[TM];

  const int e     = blockIdx.y;
  const int mbase = blockIdx.x * TM;
  const int tid   = threadIdx.x;

  const int cnt = counts[e];
  if (mbase >= cnt) return;               // uniform early-exit

  if (tid < TM) {
    int idx = mbase + tid;
    if (idx < cnt) { sTok[tid] = tokens[e * B_N + idx]; sGate[tid] = gates[e * B_N + idx]; }
    else           { sTok[tid] = -1;                    sGate[tid] = 0.f; }
  }
  __syncthreads();

  const int lane = tid & 31;
  const int wv   = tid >> 5;
  const int wm   = (wv & 1) * 32;         // wave M offset in tile
  const int wn   = (wv >> 1) * 32;        // wave N offset in tile
  const int lr   = lane & 15;             // row/col within fragment
  const int hi   = lane >> 4;             // half-wave select

  // ---------------- Phase A: h = relu(Xg @ W1 + b1) ----------------
  for (int h0 = 0; h0 < H_DIM; h0 += TN) {
    v8f acc[2][2] = {};
    for (int k0 = 0; k0 < D_DIM; k0 += TK) {
      { // stage gathered x tile (TM x TK), f32 -> f16
        int r  = tid >> 2;
        int cb = (tid & 3) * 16;
        int tok = sTok[r];
        _Float16* dst = &sA[r][cb];
        if (tok >= 0) {
          const float4* src = (const float4*)(x + (size_t)tok * D_DIM + k0 + cb);
          #pragma unroll
          for (int q = 0; q < 4; ++q) {
            float4 v = src[q];
            dst[q*4+0] = (_Float16)v.x; dst[q*4+1] = (_Float16)v.y;
            dst[q*4+2] = (_Float16)v.z; dst[q*4+3] = (_Float16)v.w;
          }
        } else {
          #pragma unroll
          for (int q = 0; q < 16; ++q) dst[q] = (_Float16)0.f;
        }
      }
      { // stage W1 chunk (TK x TN) transposed into sB[n][k]
        int k  = tid >> 2;
        int nb = (tid & 3) * 32;
        const float* src = W1 + ((size_t)e * D_DIM + (k0 + k)) * H_DIM + h0 + nb;
        #pragma unroll
        for (int q = 0; q < 8; ++q) {
          float4 v = ((const float4*)src)[q];
          int n = nb + q * 4;
          sB[n+0][k] = (_Float16)v.x; sB[n+1][k] = (_Float16)v.y;
          sB[n+2][k] = (_Float16)v.z; sB[n+3][k] = (_Float16)v.w;
        }
        if (k0 + TK < D_DIM)
          __builtin_prefetch(src + (size_t)TK * H_DIM, 0, 0);   // global_prefetch_b8
      }
      __syncthreads();
      #pragma unroll
      for (int ks = 0; ks < 2; ++ks) {
        v16h af[2], bf[2];
        #pragma unroll
        for (int fm = 0; fm < 2; ++fm) {  // A: lane<16 -> K{0..15}
          const _Float16* ap = &sA[wm + fm*16 + lr][ks*32 + hi*16];
          af[fm] = cat8(*(const v8h*)ap, *(const v8h*)(ap + 8));
        }
        #pragma unroll
        for (int fn = 0; fn < 2; ++fn)    // B: lane<16 -> K{0..15}
          bf[fn] = *(const v16h*)&sB[wn + fn*16 + lr][ks*32 + hi*16];
        #pragma unroll
        for (int fm = 0; fm < 2; ++fm)
          #pragma unroll
          for (int fn = 0; fn < 2; ++fn)
            acc[fm][fn] = __builtin_amdgcn_wmma_f32_16x16x32_f16(
                false, af[fm], false, bf[fn], (short)0, acc[fm][fn], false, false);
      }
      __syncthreads();
    }
    // epilogue: bias + relu -> LDS-resident h
    #pragma unroll
    for (int fn = 0; fn < 2; ++fn) {
      int col = h0 + wn + fn * 16 + lr;
      float bias = b1[e * H_DIM + col];
      #pragma unroll
      for (int fm = 0; fm < 2; ++fm) {
        #pragma unroll
        for (int i = 0; i < 8; ++i) {
          int row = wm + fm * 16 + hi * 8 + i;   // C layout: lanes>=16 -> M+8
          float v = acc[fm][fn][i] + bias;
          sH[row][col] = (_Float16)(v > 0.f ? v : 0.f);
        }
      }
    }
  }
  __syncthreads();

  // ---------------- Phase B: out += gate * (h @ W2 + b2) ----------------
  for (int d0 = 0; d0 < D_DIM; d0 += TN) {
    v8f acc[2][2] = {};
    for (int k0 = 0; k0 < H_DIM; k0 += TK) {
      { // stage W2 chunk (TK x TN) transposed into sB[n][k]
        int k  = tid >> 2;
        int nb = (tid & 3) * 32;
        const float* src = W2 + ((size_t)e * H_DIM + (k0 + k)) * D_DIM + d0 + nb;
        #pragma unroll
        for (int q = 0; q < 8; ++q) {
          float4 v = ((const float4*)src)[q];
          int n = nb + q * 4;
          sB[n+0][k] = (_Float16)v.x; sB[n+1][k] = (_Float16)v.y;
          sB[n+2][k] = (_Float16)v.z; sB[n+3][k] = (_Float16)v.w;
        }
        if (k0 + TK < H_DIM)
          __builtin_prefetch(src + (size_t)TK * D_DIM, 0, 0);
      }
      __syncthreads();
      #pragma unroll
      for (int ks = 0; ks < 2; ++ks) {
        v16h af[2], bf[2];
        #pragma unroll
        for (int fm = 0; fm < 2; ++fm) {  // A-fragments straight from LDS h
          const _Float16* ap = &sH[wm + fm*16 + lr][k0 + ks*32 + hi*16];
          af[fm] = cat8(*(const v8h*)ap, *(const v8h*)(ap + 8));
        }
        #pragma unroll
        for (int fn = 0; fn < 2; ++fn)
          bf[fn] = *(const v16h*)&sB[wn + fn*16 + lr][ks*32 + hi*16];
        #pragma unroll
        for (int fm = 0; fm < 2; ++fm)
          #pragma unroll
          for (int fn = 0; fn < 2; ++fn)
            acc[fm][fn] = __builtin_amdgcn_wmma_f32_16x16x32_f16(
                false, af[fm], false, bf[fn], (short)0, acc[fm][fn], false, false);
      }
      __syncthreads();
    }
    // epilogue: bias + gate, scatter-add into residual-initialized output
    #pragma unroll
    for (int fn = 0; fn < 2; ++fn) {
      int col = d0 + wn + fn * 16 + lr;
      float bias = b2[e * D_DIM + col];
      #pragma unroll
      for (int fm = 0; fm < 2; ++fm) {
        #pragma unroll
        for (int i = 0; i < 8; ++i) {
          int row = wm + fm * 16 + hi * 8 + i;
          int tok = sTok[row];
          if (tok >= 0) {
            float v = (acc[fm][fn][i] + bias) * sGate[row];
            atomicAdd(out + (size_t)tok * D_DIM + col, v);
          }
        }
      }
    }
  }
}

// ---------------------------------------------------------------------------
extern "C" void kernel_launch(void* const* d_in, const int* in_sizes, int n_in,
                              void* d_out, int out_size, void* d_ws, size_t ws_size,
                              hipStream_t stream) {
  const float* x  = (const float*)d_in[0];   // (B, D)
  const float* W1 = (const float*)d_in[1];   // (E, D, H)
  const float* b1 = (const float*)d_in[2];   // (E, H)
  const float* W2 = (const float*)d_in[3];   // (E, H, D)
  const float* b2 = (const float*)d_in[4];   // (E, D)
  const float* Wr = (const float*)d_in[5];   // (D, E)
  const float* br = (const float*)d_in[6];   // (E,)
  float* out = (float*)d_out;                // (B, D)

  // workspace carve (~0.5 MB): counters + per-expert token buckets + gates
  char* w = (char*)d_ws;
  int*   counts = (int*)w;              w += 256;
  int*   tokens = (int*)w;              w += (size_t)E_N * B_N * sizeof(int);
  float* gates  = (float*)w;            // E_N * B_N floats

  moe_zero_counts<<<1, 32, 0, stream>>>(counts);
  moe_router<<<B_N / 16, 256, 0, stream>>>(x, Wr, br, counts, tokens, gates);
  moe_residual<<<(B_N * D_DIM / 4 + 255) / 256, 256, 0, stream>>>(x, out, B_N * D_DIM / 4);
  moe_expert<<<dim3(TILES, E_N), 256, 0, stream>>>(x, W1, b1, W2, b2,
                                                   counts, tokens, gates, out);
}
